// BasicAttention_16381005267227
// MI455X (gfx1250) — matrix operand order repaired and checked by reading, not compiled, not run on hardware
//
#include <hip/hip_runtime.h>

typedef __attribute__((ext_vector_type(16))) _Float16 v16h;
typedef __attribute__((ext_vector_type(8)))  _Float16 v8h;
typedef __attribute__((ext_vector_type(4)))  _Float16 v4h;
typedef __attribute__((ext_vector_type(8)))  float    v8f;

#define B_    16
#define LQ    512
#define LC    2048
#define DD    1024
#define BM    64           // context rows per workgroup
#define NWAVE 4            // waves per workgroup (wave32)
#define STRH  520          // LDS row stride in halfs (260 dwords, %64 = 4 -> staggered)
#define NTG   8            // N-tiles per accumulator group (8 x v8f = 64 VGPRs)
#define NSLOT 3            // software-pipeline depth for B fragments

#if __has_builtin(__builtin_amdgcn_sched_barrier)
#define SCHED_FENCE() __builtin_amdgcn_sched_barrier(0)
#else
#define SCHED_FENCE()
#endif

// ---------------------------------------------------------------------------
// Kernel 0: pack question fp32 -> f16, both row-major (qh) and transposed (qt)
// ---------------------------------------------------------------------------
__global__ void pack_q(const float* __restrict__ q,
                       _Float16* __restrict__ qh,
                       _Float16* __restrict__ qt) {
    size_t idx  = (size_t)blockIdx.x * blockDim.x + threadIdx.x; // over B*LQ*DD/4
    size_t base = idx * 4;
    int d  = (int)(base % DD);
    size_t t = base / DD;
    int qq = (int)(t % LQ);
    int b  = (int)(t / LQ);

    const float4 v = *(const float4*)(q + ((size_t)b * LQ + qq) * DD + d);
    _Float16 h0 = (_Float16)v.x, h1 = (_Float16)v.y;
    _Float16 h2 = (_Float16)v.z, h3 = (_Float16)v.w;

    v4h pk; pk[0] = h0; pk[1] = h1; pk[2] = h2; pk[3] = h3;
    *(v4h*)(qh + ((size_t)b * LQ + qq) * DD + d) = pk;

    _Float16* dt = qt + ((size_t)b * DD + d) * LQ + qq;
    dt[0]        = h0;
    dt[LQ]       = h1;
    dt[2 * LQ]   = h2;
    dt[3 * LQ]   = h3;
}

// ---------------------------------------------------------------------------
// Kernel 1: S = C @ Q^T (f16 WMMA, f32 acc), row softmax, P (f16) to workspace
// ---------------------------------------------------------------------------
__global__ void qk_softmax(const float* __restrict__ ctx,
                           const _Float16* __restrict__ qh,
                           _Float16* __restrict__ P) {
    extern __shared__ _Float16 sS[];           // [BM][STRH] f16 scores

    const int b    = blockIdx.y;
    const int row0 = blockIdx.x * BM;
    const int lane = threadIdx.x & 31;
    const int wave = threadIdx.x >> 5;
    const int h    = lane >> 4;                // K-half select (ISA A/B layouts)
    const int mn   = lane & 15;                // M (for A) / N (for B) index

    const float*    arow = ctx + ((size_t)b * LC + row0 + wave * 16 + mn) * DD;
    const _Float16* qrow = qh + (size_t)b * LQ * DD + (size_t)mn * DD;

    for (int ntg = 0; ntg < LQ / 16; ntg += NTG) {
        v8f acc[NTG];
#pragma unroll
        for (int j = 0; j < NTG; ++j) acc[j] = (v8f){};

        for (int k0 = 0; k0 < DD; k0 += 32) {
            __builtin_prefetch(arow + k0 + 256, 0, 1);

            v8h s0[NSLOT], s1[NSLOT];
            // pipeline prologue: B fragments for tiles 0 and 1
#pragma unroll
            for (int j = 0; j < 2; ++j) {
                const _Float16* br = qrow + (size_t)(ntg + j) * 16 * DD;
                s0[j] = *(const v8h*)(br + k0 + 16 * h);
                s1[j] = *(const v8h*)(br + k0 + 16 * h + 8);
            }

            // --- A fragment (shared across the NTG group): fp32 -> f16 ---
            // lane half h: elems 0..15 <-> K=k0+16h+0..15
            float4 c0 = *(const float4*)(arow + k0 + 16 * h);
            float4 c1 = *(const float4*)(arow + k0 + 16 * h + 4);
            float4 c2 = *(const float4*)(arow + k0 + 16 * h + 8);
            float4 c3 = *(const float4*)(arow + k0 + 16 * h + 12);
            v16h a;
            a[0] = (_Float16)c0.x; a[1] = (_Float16)c0.y; a[2] = (_Float16)c0.z; a[3] = (_Float16)c0.w;
            a[4] = (_Float16)c1.x; a[5] = (_Float16)c1.y; a[6] = (_Float16)c1.z; a[7] = (_Float16)c1.w;
            a[8]  = (_Float16)c2.x; a[9]  = (_Float16)c2.y; a[10] = (_Float16)c2.z; a[11] = (_Float16)c2.w;
            a[12] = (_Float16)c3.x; a[13] = (_Float16)c3.y; a[14] = (_Float16)c3.z; a[15] = (_Float16)c3.w;

            // --- pipelined WMMA chain: load B(j+2), fence, WMMA(j) ---
#pragma unroll
            for (int j = 0; j < NTG; ++j) {
                if (j + 2 < NTG) {
                    const int sl = (j + 2) % NSLOT;
                    const _Float16* br = qrow + (size_t)(ntg + j + 2) * 16 * DD;
                    s0[sl] = *(const v8h*)(br + k0 + 16 * h);
                    s1[sl] = *(const v8h*)(br + k0 + 16 * h + 8);
                }
                SCHED_FENCE();
                const int cu = j % NSLOT;
                v16h bb;
#pragma unroll
                for (int i = 0; i < 8; ++i) { bb[i] = s0[cu][i]; bb[i + 8] = s1[cu][i]; }
                acc[j] = __builtin_amdgcn_wmma_f32_16x16x32_f16(
                    false, a, false, bb, (short)0, acc[j], false, false);
            }
        }
        // C layout: VGPR r -> M = r + 8h, N = lane&15 ; store as f16
#pragma unroll
        for (int j = 0; j < NTG; ++j)
#pragma unroll
            for (int r = 0; r < 8; ++r)
                sS[(size_t)(wave * 16 + r + 8 * h) * STRH + (ntg + j) * 16 + mn] =
                    (_Float16)acc[j][r];
    }
    __syncthreads();

    // per-wave softmax over its 16 rows; 32 lanes cooperate per row (fp32 math)
    for (int r = 0; r < 16; ++r) {
        const _Float16* srow = sS + (size_t)(wave * 16 + r) * STRH;
        float mx = -3.0e38f;
#pragma unroll
        for (int c = 0; c < LQ / 32; ++c) mx = fmaxf(mx, (float)srow[lane + 32 * c]);
#pragma unroll
        for (int off = 16; off; off >>= 1) mx = fmaxf(mx, __shfl_xor(mx, off, 32));

        float vals[LQ / 32];
        float sum = 0.0f;
#pragma unroll
        for (int c = 0; c < LQ / 32; ++c) {
            float e = __expf((float)srow[lane + 32 * c] - mx);
            vals[c] = e;
            sum += e;
        }
#pragma unroll
        for (int off = 16; off; off >>= 1) sum += __shfl_xor(sum, off, 32);
        float inv = 1.0f / sum;

        _Float16* prow = P + ((size_t)b * LC + row0 + wave * 16 + r) * LQ;
#pragma unroll
        for (int c = 0; c < LQ / 32; ++c)
            prow[lane + 32 * c] = (_Float16)(vals[c] * inv);
    }
}

// ---------------------------------------------------------------------------
// Kernel 2: O = P @ Q (f16 WMMA), out = [context | O]
// ---------------------------------------------------------------------------
__global__ void pv_concat(const float* __restrict__ ctx,
                          const _Float16* __restrict__ qt,
                          const _Float16* __restrict__ P,
                          float* __restrict__ out) {
    const int b    = blockIdx.y;
    const int row0 = blockIdx.x * BM;
    const int lane = threadIdx.x & 31;
    const int wave = threadIdx.x >> 5;
    const int h    = lane >> 4;
    const int mn   = lane & 15;

    // copy context rows into out[..., 0:DD] (vectorized)
    {
        const float4* src = (const float4*)(ctx + ((size_t)b * LC + row0) * DD);
        for (int i = threadIdx.x; i < BM * DD / 4; i += blockDim.x) {
            int rr = i / (DD / 4), cc = i % (DD / 4);
            float4 v = src[(size_t)rr * (DD / 4) + cc];
            *(float4*)(out + ((size_t)b * LC + row0 + rr) * (2 * DD) + cc * 4) = v;
        }
    }

    const _Float16* prow = P + ((size_t)b * LC + row0 + wave * 16 + mn) * LQ;
    const _Float16* qtb  = qt + (size_t)b * DD * LQ + (size_t)mn * LQ;

    for (int ntg = 0; ntg < DD / 16; ntg += NTG) {
        v8f acc[NTG];
#pragma unroll
        for (int j = 0; j < NTG; ++j) acc[j] = (v8f){};

        for (int k0 = 0; k0 < LQ; k0 += 32) {
            v8h s0[NSLOT], s1[NSLOT];
#pragma unroll
            for (int j = 0; j < 2; ++j) {
                const _Float16* br = qtb + (size_t)(ntg + j) * 16 * LQ;
                s0[j] = *(const v8h*)(br + k0 + 16 * h);
                s1[j] = *(const v8h*)(br + k0 + 16 * h + 8);
            }

            // A fragment from P (f16), shared across the NTG group
            v8h a0 = *(const v8h*)(prow + k0 + 16 * h);
            v8h a1 = *(const v8h*)(prow + k0 + 16 * h + 8);
            v16h a;
#pragma unroll
            for (int i = 0; i < 8; ++i) { a[i] = a0[i]; a[i + 8] = a1[i]; }

#pragma unroll
            for (int j = 0; j < NTG; ++j) {
                if (j + 2 < NTG) {
                    const int sl = (j + 2) % NSLOT;
                    const _Float16* br = qtb + (size_t)(ntg + j + 2) * 16 * LQ;
                    s0[sl] = *(const v8h*)(br + k0 + 16 * h);
                    s1[sl] = *(const v8h*)(br + k0 + 16 * h + 8);
                }
                SCHED_FENCE();
                const int cu = j % NSLOT;
                v16h bb;
#pragma unroll
                for (int i = 0; i < 8; ++i) { bb[i] = s0[cu][i]; bb[i + 8] = s1[cu][i]; }
                acc[j] = __builtin_amdgcn_wmma_f32_16x16x32_f16(
                    false, a, false, bb, (short)0, acc[j], false, false);
            }
        }
#pragma unroll
        for (int j = 0; j < NTG; ++j)
#pragma unroll
            for (int r = 0; r < 8; ++r)
                out[((size_t)b * LC + row0 + wave * 16 + r + 8 * h) * (2 * DD)
                    + DD + (ntg + j) * 16 + mn] = acc[j][r];
    }
}

// ---------------------------------------------------------------------------
extern "C" void kernel_launch(void* const* d_in, const int* in_sizes, int n_in,
                              void* d_out, int out_size, void* d_ws, size_t ws_size,
                              hipStream_t stream) {
    (void)in_sizes; (void)n_in; (void)out_size; (void)ws_size;
    const float* question = (const float*)d_in[0];
    const float* context  = (const float*)d_in[1];
    float* out = (float*)d_out;

    char* ws = (char*)d_ws;
    _Float16* qh = (_Float16*)ws;                                    // 16 MiB
    _Float16* qt = (_Float16*)(ws + (size_t)B_ * LQ * DD * 2);       // 16 MiB
    _Float16* P  = (_Float16*)(ws + (size_t)B_ * LQ * DD * 4);       // 32 MiB

    {
        size_t tot4 = (size_t)B_ * LQ * DD / 4;
        pack_q<<<dim3((unsigned)(tot4 / 256)), 256, 0, stream>>>(question, qh, qt);
    }
    qk_softmax<<<dim3(LC / BM, B_), NWAVE * 32,
                 BM * STRH * sizeof(_Float16), stream>>>(context, qh, P);
    pv_concat<<<dim3(LC / BM, B_), NWAVE * 32, 0, stream>>>(context, qt, P, out);
}
